// SimpleMoE_50869592653849
// MI455X (gfx1250) — compile-verified, matrix operand order checked by reading, not run on hardware
//
#include <hip/hip_runtime.h>
#include <hip/hip_bf16.h>
#include <math.h>

// ---------------- MoE geometry (compile-time, matches reference) -----------
#define N_EXPERTS 8
#define TOP_K     2
#define D_MODEL   1024
#define D_FF      4096
#define N_TOKENS  8192          // B*T = 4*2048
#define M_TILE    32            // tokens per workgroup (two WMMA M-subtiles)
#define F_BLOCK   128           // D_FF chunk held in LDS per iteration
#define BLOCK_THREADS 256       // 8 wave32s

// Packed-weight tile grids (32x16 K-by-N tiles, one v16h per lane per tile)
#define W1_KT (D_MODEL / 32)    // 32
#define W1_NT (D_FF / 16)       // 256
#define W2_KT (D_FF / 32)       // 128
#define W2_NT (D_MODEL / 16)    // 64

typedef __attribute__((ext_vector_type(16))) _Float16 v16h;
typedef __attribute__((ext_vector_type(8)))  _Float16 v8h;
typedef __attribute__((ext_vector_type(8)))  float    v8f;

// ---------------------------------------------------------------------------
// A-fragment (16xK tile, K-step of 32) from LDS f16, row-major, stride in halves.
// ISA layout: lanes 0-15 -> row M=lane, halves K = ks*32 + {0..7,16..23};
//             lanes 16-31 -> row M=lane-16, halves K = ks*32 + {8..15,24..31}.
__device__ __forceinline__ v16h load_a_lds(const _Float16* base, int strideH,
                                           int kbase, int lane) {
  const int row = lane & 15;
  const int hi  = (lane >> 4) << 3;          // 0 or 8
  const _Float16* p = base + row * strideH + kbase + hi;
  v8h lo = *(const v8h*)(p);
  v8h up = *(const v8h*)(p + 16);
  return __builtin_shufflevector(lo, up, 0,1,2,3,4,5,6,7,8,9,10,11,12,13,14,15);
}

// Fallback B-fragment (32x16 tile) gathered from a row-major f32 matrix.
__device__ __forceinline__ v16h load_b_g(const float* w, int ldw,
                                         int krow0, int col0, int lane) {
  const int hi = (lane >> 4) << 3;
  const float* p = w + (size_t)(krow0 + hi) * ldw + (col0 + (lane & 15));
  v16h b;
#pragma unroll
  for (int j = 0; j < 8; ++j) {
    b[j]     = (_Float16)p[(size_t)j * ldw];
    b[j + 8] = (_Float16)p[(size_t)(j + 16) * ldw];
  }
  return b;
}

__device__ __forceinline__ v8f wmma_f16(v16h a, v16h b, v8f c) {
  return __builtin_amdgcn_wmma_f32_16x16x32_f16(false, a, false, b,
                                                (short)0, c, false, false);
}

// ---------------------------------------------------------------------------
// One-time weight pack: row-major f32 W[e][K][N] -> per-tile per-lane f16
// B-fragment images. Tile t covers K rows [kt*32, kt*32+32), cols [nt*16,+16).
__global__ __launch_bounds__(BLOCK_THREADS)
void pack_weights_kernel(const float* __restrict__ w, _Float16* __restrict__ wp,
                         int K, int N) {
  const int lane = threadIdx.x & 31;
  const int tile = blockIdx.x * (BLOCK_THREADS / 32) + (threadIdx.x >> 5);
  const int Kt = K / 32, Nt = N / 16;
  const int tilesPerExpert = Kt * Nt;
  if (tile >= N_EXPERTS * tilesPerExpert) return;
  const int e  = tile / tilesPerExpert;
  const int r  = tile - e * tilesPerExpert;
  const int kt = r / Nt;
  const int nt = r - kt * Nt;

  const float* we = w + (size_t)e * K * N;
  const int hi  = (lane >> 4) << 3;
  const int col = nt * 16 + (lane & 15);
  const int k0  = kt * 32 + hi;

  v16h b;
#pragma unroll
  for (int j = 0; j < 8; ++j) {
    b[j]     = (_Float16)we[(size_t)(k0 + j) * N + col];
    b[j + 8] = (_Float16)we[(size_t)(k0 + 16 + j) * N + col];
  }
  ((v16h*)wp)[(size_t)tile * 32 + lane] = b;
}

// ---------------------------------------------------------------------------
// Router: one wave32 per token. logits = x[t]·gate_w + gate_b, top-2,
// softmax over the two, scatter token into its two expert buckets.
__global__ __launch_bounds__(BLOCK_THREADS)
void moe_router_kernel(const float* __restrict__ x,
                       const float* __restrict__ gate_w,
                       const float* __restrict__ gate_b,
                       float* __restrict__ coef,      // [N_TOKENS][N_EXPERTS]
                       int*   __restrict__ idxbuf,    // [N_EXPERTS][N_TOKENS]
                       int*   __restrict__ cnts) {    // [N_EXPERTS]
  const int lane = threadIdx.x & 31;
  const int wv   = threadIdx.x >> 5;
  const int t    = blockIdx.x * (BLOCK_THREADS / 32) + wv;
  if (t >= N_TOKENS) return;

  float acc[N_EXPERTS];
#pragma unroll
  for (int e = 0; e < N_EXPERTS; ++e) acc[e] = 0.0f;

  const float* xr = x + (size_t)t * D_MODEL;
#pragma unroll 4
  for (int i = 0; i < D_MODEL / 32; ++i) {
    const int d = lane + i * 32;
    const float xv = xr[d];
    const float* g = gate_w + d * N_EXPERTS;
#pragma unroll
    for (int e = 0; e < N_EXPERTS; ++e) acc[e] += xv * g[e];
  }
#pragma unroll
  for (int e = 0; e < N_EXPERTS; ++e) {
#pragma unroll
    for (int off = 16; off > 0; off >>= 1)
      acc[e] += __shfl_down(acc[e], off, 32);
  }

  if (lane == 0) {
    float v[N_EXPERTS];
#pragma unroll
    for (int e = 0; e < N_EXPERTS; ++e) v[e] = acc[e] + gate_b[e];
    int i0 = 0;
#pragma unroll
    for (int e = 1; e < N_EXPERTS; ++e) if (v[e] > v[i0]) i0 = e;
    int i1 = (i0 == 0) ? 1 : 0;
#pragma unroll
    for (int e = 0; e < N_EXPERTS; ++e)
      if (e != i0 && v[e] > v[i1]) i1 = e;
    const float m  = fmaxf(v[i0], v[i1]);
    const float e0 = __expf(v[i0] - m);
    const float e1 = __expf(v[i1] - m);
    const float inv = 1.0f / (e0 + e1);
    float* cr = coef + (size_t)t * N_EXPERTS;
#pragma unroll
    for (int e = 0; e < N_EXPERTS; ++e) cr[e] = 0.0f;
    cr[i0] = e0 * inv;
    cr[i1] = e1 * inv;
    int p0 = atomicAdd(&cnts[i0], 1);
    idxbuf[i0 * N_TOKENS + p0] = t;
    int p1 = atomicAdd(&cnts[i1], 1);
    idxbuf[i1 * N_TOKENS + p1] = t;
  }
}

// ---------------------------------------------------------------------------
// Fused expert FFN over gathered tokens, M_TILE = 32 (two WMMA M-subtiles so
// every B fragment feeds two WMMAs -> half the L2 operand traffic per FLOP).
// grid = (N_TOKENS/M_TILE, N_EXPERTS); block = 256 (8 waves).
// Wave w: FFN1 -> h columns [f0 + 16w, f0 + 16w + 16), rows 0..31
//         FFN2 -> y columns [128w, 128w + 128) (8 n-tiles x 2 m-subtiles)
// Dynamic LDS: x tile 32x1024 f16 (64KB) + h tile 32x128 f16 (8KB).
template <bool PACKED>
__global__ __launch_bounds__(BLOCK_THREADS, 1)
void moe_ffn_kernel(const float* __restrict__ x,
                    const float* __restrict__ w1,
                    const float* __restrict__ b1,
                    const float* __restrict__ w2,
                    const float* __restrict__ b2,
                    const _Float16* __restrict__ w1p,
                    const _Float16* __restrict__ w2p,
                    const float* __restrict__ coef,
                    const int*   __restrict__ idxbuf,
                    const int*   __restrict__ cnts,
                    float*       __restrict__ out) {
  extern __shared__ __align__(16) _Float16 smem[];
  _Float16* x_lds = smem;                          // M_TILE * D_MODEL halves
  _Float16* h_lds = smem + M_TILE * D_MODEL;       // M_TILE * F_BLOCK halves
  __shared__ int   tokS[M_TILE];
  __shared__ float coefS[M_TILE];

  const int e   = blockIdx.y;
  const int cnt = cnts[e];
  const int m0  = blockIdx.x * M_TILE;
  if (m0 >= cnt) return;                       // uniform early exit

  const int tid  = threadIdx.x;
  const int lane = tid & 31;
  const int wv   = tid >> 5;

  if (tid < M_TILE) {
    const int r = m0 + tid;
    const int tok = (r < cnt) ? idxbuf[e * N_TOKENS + r] : 0;
    tokS[tid]  = tok;
    coefS[tid] = (r < cnt) ? coef[(size_t)tok * N_EXPERTS + e] : 0.0f;
  }
  __syncthreads();

  // Stage x tile (f32 -> f16) into LDS; coalesced over k.
#pragma unroll 4
  for (int i = 0; i < (M_TILE * D_MODEL) / BLOCK_THREADS; ++i) {
    const int flat = tid + i * BLOCK_THREADS;
    const int row = flat >> 10;
    const int k   = flat & (D_MODEL - 1);
    x_lds[row * D_MODEL + k] = (_Float16)x[(size_t)tokS[row] * D_MODEL + k];
  }
  __syncthreads();

  const float* w1e = w1 + (size_t)e * D_MODEL * D_FF;   // [D_MODEL][D_FF]
  const float* w2e = w2 + (size_t)e * D_FF * D_MODEL;   // [D_FF][D_MODEL]
  const v16h*  w1pv = (const v16h*)w1p;
  const v16h*  w2pv = (const v16h*)w2p;
  const size_t w1base = (size_t)e * W1_KT * W1_NT;      // expert tile base
  const size_t w2base = (size_t)e * W2_KT * W2_NT;

  v8f yacc[2][8];                              // [m-subtile][n-tile]
#pragma unroll
  for (int ms = 0; ms < 2; ++ms)
#pragma unroll
    for (int nt = 0; nt < 8; ++nt) yacc[ms][nt] = (v8f){};

  const int ncol0 = wv * 128;
  const int mrow0 = (lane >> 4) << 3;

#pragma unroll 1
  for (int fb = 0; fb < D_FF / F_BLOCK; ++fb) {
    const int f0 = fb * F_BLOCK;
    if (fb + 1 < D_FF / F_BLOCK) {             // hint next f-block into cache
      if constexpr (PACKED)
        __builtin_prefetch(
            (const void*)&w1pv[(w1base + (fb + 1) * 8 + wv) * 32 + lane], 0, 1);
      else
        __builtin_prefetch(w1e + (size_t)lane * D_FF + f0 + F_BLOCK, 0, 1);
    }

    // ---- FFN1: h[32 x 16] for this wave's columns, K = 1024 ----
    const int hcol = f0 + wv * 16;             // -> n-tile index fb*8 + wv
    v8f hacc[2] = {(v8f){}, (v8f){}};
#pragma unroll 1
    for (int ks = 0; ks < D_MODEL / 32; ++ks) {
      v16h b;
      if constexpr (PACKED)
        b = w1pv[(w1base + (size_t)ks * W1_NT + (fb * 8 + wv)) * 32 + lane];
      else
        b = load_b_g(w1e, D_FF, ks * 32, hcol, lane);
      v16h a0 = load_a_lds(x_lds, D_MODEL, ks * 32, lane);
      v16h a1 = load_a_lds(x_lds + 16 * D_MODEL, D_MODEL, ks * 32, lane);
      hacc[0] = wmma_f16(a0, b, hacc[0]);      // B reused twice
      hacc[1] = wmma_f16(a1, b, hacc[1]);
    }
    // bias + exact GELU, store f16 h block to LDS
    const int   nloc = (lane & 15);
    const float bias = b1[e * D_FF + hcol + nloc];
#pragma unroll
    for (int ms = 0; ms < 2; ++ms) {
#pragma unroll
      for (int v = 0; v < 8; ++v) {
        const float hv = hacc[ms][v] + bias;
        const float g  = 0.5f * hv * (1.0f + erff(hv * 0.70710678118654752f));
        h_lds[(ms * 16 + mrow0 + v) * F_BLOCK + wv * 16 + nloc] = (_Float16)g;
      }
    }
    __syncthreads();

    // ---- FFN2 partial: y += h_block @ w2[f0:f0+128, :], K = 128 ----
#pragma unroll
    for (int ks = 0; ks < F_BLOCK / 32; ++ks) {
      v16h a0 = load_a_lds(h_lds, F_BLOCK, ks * 32, lane);
      v16h a1 = load_a_lds(h_lds + 16 * F_BLOCK, F_BLOCK, ks * 32, lane);
      const int kt2 = fb * 4 + ks;             // w2 K-tile index
#pragma unroll
      for (int nt = 0; nt < 8; ++nt) {
        v16h b2f;
        if constexpr (PACKED)
          b2f = w2pv[(w2base + (size_t)kt2 * W2_NT + (wv * 8 + nt)) * 32 + lane];
        else
          b2f = load_b_g(w2e, D_MODEL, f0 + ks * 32, ncol0 + nt * 16, lane);
        yacc[0][nt] = wmma_f16(a0, b2f, yacc[0][nt]);  // B reused twice
        yacc[1][nt] = wmma_f16(a1, b2f, yacc[1][nt]);
      }
    }
    __syncthreads();                           // h_lds reused next fb
  }

  // ---- Epilogue: out[tok] += coef * (y + b2) ----
#pragma unroll
  for (int ms = 0; ms < 2; ++ms) {
#pragma unroll
    for (int nt = 0; nt < 8; ++nt) {
      const int n  = ncol0 + nt * 16 + (lane & 15);
      const float bv = b2[e * D_MODEL + n];
#pragma unroll
      for (int v = 0; v < 8; ++v) {
        const int m = ms * 16 + mrow0 + v;
        if (m0 + m < cnt) {
          atomicAdd(&out[(size_t)tokS[m] * D_MODEL + n],
                    coefS[m] * (yacc[ms][nt][v] + bv));
        }
      }
    }
  }
}

// ---------------------------------------------------------------------------
extern "C" void kernel_launch(void* const* d_in, const int* in_sizes, int n_in,
                              void* d_out, int out_size, void* d_ws, size_t ws_size,
                              hipStream_t stream) {
  const float* x      = (const float*)d_in[0];
  const float* gate_w = (const float*)d_in[1];
  const float* gate_b = (const float*)d_in[2];
  const float* w1     = (const float*)d_in[3];
  const float* b1     = (const float*)d_in[4];
  const float* w2     = (const float*)d_in[5];
  const float* b2     = (const float*)d_in[6];
  float* out = (float*)d_out;

  // Workspace layout (all offsets 256B-aligned):
  //   [0,256)            expert counters
  //   [256, +256KB)      coef
  //   [.., +256KB)       idx buckets
  //   [.., +64MB)        packed f16 w1 (optional)
  //   [.., +64MB)        packed f16 w2 (optional)
  char* ws = (char*)d_ws;
  int*      cnts   = (int*)ws;
  float*    coef   = (float*)(ws + 256);
  int*      idxbuf = (int*)(ws + 256 + (size_t)N_TOKENS * N_EXPERTS * sizeof(float));
  size_t    off    = 256 + 2ull * N_TOKENS * N_EXPERTS * sizeof(float); // 524544
  _Float16* w1p    = (_Float16*)(ws + off);
  const size_t w1pBytes = (size_t)N_EXPERTS * D_MODEL * D_FF * sizeof(_Float16);
  _Float16* w2p    = (_Float16*)(ws + off + w1pBytes);
  const size_t needed = off + 2 * w1pBytes;
  const bool packed = (ws_size >= needed);

  hipMemsetAsync(cnts, 0, 256, stream);
  hipMemsetAsync(out, 0, (size_t)out_size * sizeof(float), stream);

  moe_router_kernel<<<N_TOKENS / (BLOCK_THREADS / 32), BLOCK_THREADS, 0, stream>>>(
      x, gate_w, gate_b, coef, idxbuf, cnts);

  dim3 grid(N_TOKENS / M_TILE, N_EXPERTS);
  const size_t ldsBytes = (size_t)(M_TILE * D_MODEL + M_TILE * F_BLOCK)
                          * sizeof(_Float16);          // 72 KB dynamic LDS
  if (packed) {
    const int tiles1 = N_EXPERTS * W1_KT * W1_NT;     // 65536
    const int tiles2 = N_EXPERTS * W2_KT * W2_NT;     // 65536
    pack_weights_kernel<<<tiles1 / (BLOCK_THREADS / 32), BLOCK_THREADS, 0, stream>>>(
        w1, w1p, D_MODEL, D_FF);
    pack_weights_kernel<<<tiles2 / (BLOCK_THREADS / 32), BLOCK_THREADS, 0, stream>>>(
        w2, w2p, D_FF, D_MODEL);
    moe_ffn_kernel<true><<<grid, BLOCK_THREADS, ldsBytes, stream>>>(
        x, w1, b1, w2, b2, w1p, w2p, coef, idxbuf, cnts, out);
  } else {
    moe_ffn_kernel<false><<<grid, BLOCK_THREADS, ldsBytes, stream>>>(
        x, w1, b1, w2, b2, w1p, w2p, coef, idxbuf, cnts, out);
  }
}
